// Cons_15504831939191
// MI455X (gfx1250) — matrix operand order repaired and checked by reading, not compiled, not run on hardware
//
#include <hip/hip_runtime.h>
#include <hip/hip_bf16.h>

typedef _Float16 v16h __attribute__((ext_vector_type(16)));
typedef _Float16 v8h  __attribute__((ext_vector_type(8)));
typedef float    v8f  __attribute__((ext_vector_type(8)));
typedef int      v4i  __attribute__((ext_vector_type(4)));

constexpr int BN    = 4096;   // batch
constexpr int D     = 1024;   // 32*32 flattened
constexpr int NCLS  = 10;
constexpr int BT    = 128;    // block tile (rows/cols)
constexpr int KC    = 32;     // K chunk (WMMA K)
constexpr int NK    = D / KC; // 32 K chunks
constexpr int LDSTR = 40;     // LDS row stride in halves (pad vs 32)
constexpr int NB    = BN / BT;            // 32 block rows/cols
constexpr int NBLK  = NB * (NB + 1) / 2;  // 528 upper-tri blocks

#if defined(__gfx1250__) && \
    __has_builtin(__builtin_amdgcn_global_load_async_to_lds_b128) && \
    __has_builtin(__builtin_amdgcn_s_wait_asynccnt)
#define USE_ASYNC_LDS 1
#else
#define USE_ASYNC_LDS 0
#endif

// ---------------- phase 1: normalize rows, write f16 unit vectors + class ----

__device__ __forceinline__ float blk_min(float v, float* red, int t) {
    red[t] = v; __syncthreads();
    for (int s = 128; s > 0; s >>= 1) {
        if (t < s) red[t] = fminf(red[t], red[t + s]);
        __syncthreads();
    }
    float r = red[0]; __syncthreads(); return r;
}
__device__ __forceinline__ float blk_max(float v, float* red, int t) {
    red[t] = v; __syncthreads();
    for (int s = 128; s > 0; s >>= 1) {
        if (t < s) red[t] = fmaxf(red[t], red[t + s]);
        __syncthreads();
    }
    float r = red[0]; __syncthreads(); return r;
}
__device__ __forceinline__ float blk_sum(float v, float* red, int t) {
    red[t] = v; __syncthreads();
    for (int s = 128; s > 0; s >>= 1) {
        if (t < s) red[t] += red[t + s];
        __syncthreads();
    }
    float r = red[0]; __syncthreads(); return r;
}

__global__ __launch_bounds__(256) void prep_kernel(
    const float* __restrict__ outputs,   // [BN, 10]
    const float* __restrict__ grad,      // [BN, D]
    const float* __restrict__ x,         // [BN, D]
    _Float16* __restrict__ ngh,          // [BN, D] unit min-max-normalized grad
    _Float16* __restrict__ xvh,          // [BN, D] unit x
    int* __restrict__ cls)               // [BN]
{
    __shared__ float red[256];
    const int row = blockIdx.x;
    const int t   = threadIdx.x;
    const size_t base = (size_t)row * D;

    // ---- grad row: min/max normalize then L2 normalize ----
    float gv[4];
#pragma unroll
    for (int i = 0; i < 4; ++i) gv[i] = grad[base + t + 256 * i];

    float lmin = fminf(fminf(gv[0], gv[1]), fminf(gv[2], gv[3]));
    float lmax = fmaxf(fmaxf(gv[0], gv[1]), fmaxf(gv[2], gv[3]));
    float gmin = blk_min(lmin, red, t);
    float gmax = blk_max(lmax, red, t);
    float sc   = 1.0f / (gmax - gmin);

    float ng[4]; float ss = 0.0f;
#pragma unroll
    for (int i = 0; i < 4; ++i) { ng[i] = (gv[i] - gmin) * sc; ss += ng[i] * ng[i]; }
    float tot = blk_sum(ss, red, t);
    float rn  = rsqrtf(fmaxf(tot, 1e-24f));
#pragma unroll
    for (int i = 0; i < 4; ++i) ngh[base + t + 256 * i] = (_Float16)(ng[i] * rn);

    // ---- x row: L2 normalize ----
    float xvr[4]; float sx = 0.0f;
#pragma unroll
    for (int i = 0; i < 4; ++i) { xvr[i] = x[base + t + 256 * i]; sx += xvr[i] * xvr[i]; }
    float totx = blk_sum(sx, red, t);
    float rnx  = rsqrtf(fmaxf(totx, 1e-24f));
#pragma unroll
    for (int i = 0; i < 4; ++i) xvh[base + t + 256 * i] = (_Float16)(xvr[i] * rnx);

    // ---- class = argmax over 10 logits ----
    if (t == 0) {
        const float* o = outputs + (size_t)row * NCLS;
        int best = 0; float bv = o[0];
#pragma unroll
        for (int c = 1; c < NCLS; ++c) { float v = o[c]; if (v > bv) { bv = v; best = c; } }
        cls[row] = best;
    }
}

// ---------------- phase 2: fused WMMA Gram tiles + masked reduction ----------

__device__ __forceinline__ v16h load_frag_a(const _Float16* tile, int lane) {
    // A 16x32 f16: lane m = lane&15; K-halves [kh..kh+7] and [16+kh..16+kh+7], kh = (lane>=16)?8:0
    int m  = lane & 15;
    int kh = (lane >> 4) * 8;
    const v8h* p0 = reinterpret_cast<const v8h*>(tile + m * LDSTR + kh);
    const v8h* p1 = reinterpret_cast<const v8h*>(tile + m * LDSTR + 16 + kh);
    v8h lo = *p0, hi = *p1;
    return __builtin_shufflevector(lo, hi, 0,1,2,3,4,5,6,7,8,9,10,11,12,13,14,15);
}

__device__ __forceinline__ v16h load_frag_b(const _Float16* tile, int lane) {
    // B 32x16 f16 (= rows of the J-tile): lane n = lane&15; K-halves [kh..kh+7] and [16+kh..16+kh+7], kh = (lane>=16)?8:0
    int n  = lane & 15;
    int k0 = (lane >> 4) * 8;
    const v8h* p0 = reinterpret_cast<const v8h*>(tile + n * LDSTR + k0);
    const v8h* p1 = reinterpret_cast<const v8h*>(tile + n * LDSTR + 16 + k0);
    v8h lo = *p0, hi = *p1;
    return __builtin_shufflevector(lo, hi, 0,1,2,3,4,5,6,7,8,9,10,11,12,13,14,15);
}

#if USE_ASYNC_LDS
// builtin signature (from clang diagnostic): arg0 = int4 addrspace(1)* (global),
// arg1 = int4 addrspace(3)* (LDS), arg2 = imm offset, arg3 = imm cpol
typedef __attribute__((address_space(1))) v4i* gv4i_p;
typedef __attribute__((address_space(3))) v4i* lv4i_p;
// copy 32 bytes (16 halves) global -> LDS via the async-to-LDS engine (ASYNCcnt)
__device__ __forceinline__ void async_cp32(const _Float16* g, _Float16* l) {
    gv4i_p gp = (gv4i_p)g;
    lv4i_p lp = (lv4i_p)l;
    __builtin_amdgcn_global_load_async_to_lds_b128(gp,     lp,     0, 0);
    __builtin_amdgcn_global_load_async_to_lds_b128(gp + 1, lp + 1, 0, 0);
}
#endif

// stage one 32-wide K chunk of all four 128-row tiles into LDS
__device__ __forceinline__ void stage_chunk(
    const _Float16* __restrict__ ngh, const _Float16* __restrict__ xvh,
    _Float16* Asg, _Float16* Bsg, _Float16* Asx, _Float16* Bsx,
    int rowBase, int colBase, int r, int seg, int kc)
{
    const size_t koff = (size_t)kc * KC + seg * 16;
    const _Float16* sAg = ngh + (size_t)(rowBase + r) * D + koff;
    const _Float16* sBg = ngh + (size_t)(colBase + r) * D + koff;
    const _Float16* sAx = xvh + (size_t)(rowBase + r) * D + koff;
    const _Float16* sBx = xvh + (size_t)(colBase + r) * D + koff;
    const int ldst = r * LDSTR + seg * 16;
#if USE_ASYNC_LDS
    async_cp32(sAg, Asg + ldst);
    async_cp32(sBg, Bsg + ldst);
    async_cp32(sAx, Asx + ldst);
    async_cp32(sBx, Bsx + ldst);
#else
    {
        const v8h* a = reinterpret_cast<const v8h*>(sAg);
        const v8h* b = reinterpret_cast<const v8h*>(sBg);
        const v8h* c = reinterpret_cast<const v8h*>(sAx);
        const v8h* d = reinterpret_cast<const v8h*>(sBx);
        v8h a0 = a[0], a1 = a[1], b0 = b[0], b1 = b[1];
        v8h c0 = c[0], c1 = c[1], d0 = d[0], d1 = d[1];
        v8h* dA = reinterpret_cast<v8h*>(Asg + ldst);
        v8h* dB = reinterpret_cast<v8h*>(Bsg + ldst);
        v8h* dC = reinterpret_cast<v8h*>(Asx + ldst);
        v8h* dD = reinterpret_cast<v8h*>(Bsx + ldst);
        dA[0] = a0; dA[1] = a1; dB[0] = b0; dB[1] = b1;
        dC[0] = c0; dC[1] = c1; dD[0] = d0; dD[1] = d1;
    }
    if (kc + 1 < NK) {
        __builtin_prefetch(sAg + KC, 0, 3);
        __builtin_prefetch(sBx + KC, 0, 3);
    }
#endif
}

__global__ __launch_bounds__(256) void cons_main_kernel(
    const _Float16* __restrict__ ngh,
    const _Float16* __restrict__ xvh,
    const int* __restrict__ cls,
    float* __restrict__ out)
{
    // double-buffered tiles: 2 * 4 * 128 * 40 halves = 80 KB (<< 320 KB/WGP)
    __shared__ __align__(16) _Float16 Asg[2][BT * LDSTR];
    __shared__ __align__(16) _Float16 Bsg[2][BT * LDSTR];
    __shared__ __align__(16) _Float16 Asx[2][BT * LDSTR];
    __shared__ __align__(16) _Float16 Bsx[2][BT * LDSTR];
    __shared__ float partial[8];

    const int tid  = threadIdx.x;
    const int wave = tid >> 5;
    const int lane = tid & 31;

    // triangular block index -> (bi, bj), bi <= bj
    int b = blockIdx.x, bi = 0, rowlen = NB;
    while (b >= rowlen) { b -= rowlen; ++bi; --rowlen; }
    const int bj = bi + b;

    const int rowBase = bi * BT;   // global i base (A tiles)
    const int colBase = bj * BT;   // global j base (B tiles)

    v8f accG[8], accX[8];
#pragma unroll
    for (int j = 0; j < 8; ++j) {
        accG[j] = (v8f){0.f,0.f,0.f,0.f,0.f,0.f,0.f,0.f};
        accX[j] = (v8f){0.f,0.f,0.f,0.f,0.f,0.f,0.f,0.f};
    }

    const int r   = tid >> 1;      // 0..127 : row within tile for staging
    const int seg = tid & 1;       // which 16-half segment of the 32-wide K chunk

    // prologue: request chunk 0 into buffer 0
    stage_chunk(ngh, xvh, Asg[0], Bsg[0], Asx[0], Bsx[0],
                rowBase, colBase, r, seg, 0);

    for (int kc = 0; kc < NK - 1; ++kc) {
        const int cur = kc & 1;
        const int nxt = cur ^ 1;
        // request next chunk into the other buffer (safe: end-of-iter barrier
        // of the previous iteration guarantees no wave still reads it)
        stage_chunk(ngh, xvh, Asg[nxt], Bsg[nxt], Asx[nxt], Bsx[nxt],
                    rowBase, colBase, r, seg, kc + 1);
#if USE_ASYNC_LDS
        // 8 async loads per chunk per thread; loads complete in order, so
        // <=8 outstanding means chunk kc has fully landed in LDS.
        __builtin_amdgcn_s_wait_asynccnt(8);
#endif
        __syncthreads();

        const v16h ag = load_frag_a(Asg[cur] + wave * 16 * LDSTR, lane);
        const v16h ax = load_frag_a(Asx[cur] + wave * 16 * LDSTR, lane);
#pragma unroll
        for (int j = 0; j < 8; ++j) {
            v16h bg = load_frag_b(Bsg[cur] + j * 16 * LDSTR, lane);
            accG[j] = __builtin_amdgcn_wmma_f32_16x16x32_f16(
                false, ag, false, bg, (short)0, accG[j], false, false);
            v16h bx = load_frag_b(Bsx[cur] + j * 16 * LDSTR, lane);
            accX[j] = __builtin_amdgcn_wmma_f32_16x16x32_f16(
                false, ax, false, bx, (short)0, accX[j], false, false);
        }
        __syncthreads();
    }
    // peeled last chunk (immediate for s_wait_asynccnt must be an ICE)
    {
        const int cur = (NK - 1) & 1;
#if USE_ASYNC_LDS
        __builtin_amdgcn_s_wait_asynccnt(0);
#endif
        __syncthreads();
        const v16h ag = load_frag_a(Asg[cur] + wave * 16 * LDSTR, lane);
        const v16h ax = load_frag_a(Asx[cur] + wave * 16 * LDSTR, lane);
#pragma unroll
        for (int j = 0; j < 8; ++j) {
            v16h bg = load_frag_b(Bsg[cur] + j * 16 * LDSTR, lane);
            accG[j] = __builtin_amdgcn_wmma_f32_16x16x32_f16(
                false, ag, false, bg, (short)0, accG[j], false, false);
            v16h bx = load_frag_b(Bsx[cur] + j * 16 * LDSTR, lane);
            accX[j] = __builtin_amdgcn_wmma_f32_16x16x32_f16(
                false, ax, false, bx, (short)0, accX[j], false, false);
        }
    }

    // ---- epilogue: mask + ratio + reduce ----
    // C layout: VGPR rr, lanes 0-15 -> M=rr, N=lane; lanes 16-31 -> M=rr+8, N=lane-16
    const int mOff = (lane >> 4) * 8;
    const int nLoc = lane & 15;
    const int iBase = rowBase + wave * 16 + mOff;

    int clsRow[8];
#pragma unroll
    for (int rr = 0; rr < 8; ++rr) clsRow[rr] = cls[iBase + rr];

    float lsum = 0.0f;
#pragma unroll
    for (int j = 0; j < 8; ++j) {
        const int jg = colBase + j * 16 + nLoc;
        const int cj = cls[jg];
#pragma unroll
        for (int rr = 0; rr < 8; ++rr) {
            const int ig = iBase + rr;
            if (ig < jg && clsRow[rr] == cj) {
                lsum += (1.0f - accG[j][rr]) / (1.0f - accX[j][rr]);
            }
        }
    }

    // wave32 reduction
#pragma unroll
    for (int off = 16; off > 0; off >>= 1) lsum += __shfl_xor(lsum, off, 32);
    if (lane == 0) partial[wave] = lsum;
    __syncthreads();
    if (tid == 0) {
        float s = 0.0f;
#pragma unroll
        for (int w = 0; w < 8; ++w) s += partial[w];
        atomicAdd(out, s * (1.0f / (float)BN));
    }
}

// ---------------- launcher ---------------------------------------------------

extern "C" void kernel_launch(void* const* d_in, const int* in_sizes, int n_in,
                              void* d_out, int out_size, void* d_ws, size_t ws_size,
                              hipStream_t stream) {
    const float* outputs = (const float*)d_in[0];   // [4096,10]
    const float* grad    = (const float*)d_in[1];   // [4096,1,32,32]
    const float* x       = (const float*)d_in[2];   // [4096,1,32,32]
    // d_in[3] = y (int64) used only for the 1/B divisor, folded as constant

    // workspace layout
    char* ws = (char*)d_ws;
    _Float16* ngh = (_Float16*)ws;                                  // 8 MB
    _Float16* xvh = (_Float16*)(ws + (size_t)BN * D * sizeof(_Float16));
    int*      cls = (int*)(ws + 2 * (size_t)BN * D * sizeof(_Float16));

    (void)hipMemsetAsync(d_out, 0, sizeof(float), stream);

    prep_kernel<<<BN, 256, 0, stream>>>(outputs, grad, x, ngh, xvh, cls);
    cons_main_kernel<<<NBLK, 256, 0, stream>>>(ngh, xvh, cls, (float*)d_out);
}
